// MultiHeadAttention_29746943492513
// MI455X (gfx1250) — compile-verified, matrix operand order checked by reading, not run on hardware
//
#include <hip/hip_runtime.h>

// ---------------------------------------------------------------------------
// Types for CDNA5 WMMA (wave32, v_wmma_f32_16x16x32_bf16)
// ---------------------------------------------------------------------------
typedef __attribute__((ext_vector_type(16))) __bf16 v16bf;
typedef __attribute__((ext_vector_type(8)))  __bf16 v8bf;
typedef __attribute__((ext_vector_type(4)))  __bf16 v4bf;
typedef __attribute__((ext_vector_type(8)))  float  v8f;
typedef __attribute__((ext_vector_type(4)))  float  v4f;

union Frag16 { v16bf v; v8bf h[2]; };

// Problem constants (reference: B=2, S=2048, D=1024, H=16, Dh=64)
#define BB   2
#define SS   2048
#define DD   1024
#define HH   16
#define DH   64
#define MROWS (BB*SS)          // 4096
#define ATT_SCALE 0.125f       // 64^-0.5

// ---------------------------------------------------------------------------
// Async global -> LDS copy of 16 bytes per lane (CDNA5 VGLOBAL opcode 98,
// tracked by ASYNCcnt). dst must be an LDS address (low 32 bits of the
// generic pointer), src a 64-bit global address. Producer must
// s_wait_asynccnt 0 before the workgroup barrier.
// ---------------------------------------------------------------------------
__device__ __forceinline__ void async_copy_b128(void* lds_dst, const void* gsrc) {
    unsigned lds = (unsigned)(size_t)lds_dst;
    unsigned long long ga = (unsigned long long)(size_t)gsrc;
    asm volatile("global_load_async_to_lds_b128 %0, %1, off"
                 :: "v"(lds), "v"(ga) : "memory");
}
__device__ __forceinline__ void async_wait_all() {
    asm volatile("s_wait_asynccnt 0x0" ::: "memory");
}

// ---------------------------------------------------------------------------
// A/B fragment load from LDS (16x32 bf16 tile, row-major, K contiguous).
// ISA 7.12.2 (16-bit A 16x32): lanes 0-15 row m hold K {0..7,16..23},
// lanes 16-31 row m hold K {8..15,24..31} -> two ds_load_b128 per lane.
// B (32x16, K-major per column) mirrors the layout, so loading the
// N-row-major (K contiguous) tile with the same pattern yields the B frag.
// ---------------------------------------------------------------------------
__device__ __forceinline__ v16bf load_frag(const __bf16* p, int ldm, int lane) {
    const int half = lane >> 4;
    const int m    = lane & 15;
    const __bf16* r = p + m * ldm + 8 * half;
    Frag16 f;
    f.h[0] = *(const v8bf*)(r);
    f.h[1] = *(const v8bf*)(r + 16);
    return f.v;
}

__device__ __forceinline__ v8f wmma_bf16(v16bf a, v16bf b, v8f c) {
    return __builtin_amdgcn_wmma_f32_16x16x32_bf16(
        /*neg_a=*/false, a, /*neg_b=*/false, b,
        /*c_mod=*/(short)0, c, /*reuse_a=*/false, /*reuse_b=*/false);
}

// ---------------------------------------------------------------------------
// f32 -> bf16 conversion (vectorized: 4 floats -> 4 packed bf16 per step)
// ---------------------------------------------------------------------------
__global__ void cvt_f32_bf16(const float* __restrict__ x, __bf16* __restrict__ y, int n4) {
    int i = blockIdx.x * blockDim.x + threadIdx.x;
    const int stride = gridDim.x * blockDim.x;
    for (; i < n4; i += stride) {
        v4f f = *(const v4f*)(x + 4 * (size_t)i);
        v4bf b;
        b[0] = (__bf16)f[0]; b[1] = (__bf16)f[1];
        b[2] = (__bf16)f[2]; b[3] = (__bf16)f[3];
        *(v4bf*)(y + 4 * (size_t)i) = b;
    }
}

// ---------------------------------------------------------------------------
// GEMM: C[M,N] = A[M,K] * W[N,K]^T  (nn.Linear, bias-free)
// A, W bf16 row-major (K contiguous). OutT = float or __bf16 (templated:
// straight-line epilogue, no runtime branch).
// Block: 256 thr = 8 waves; tile 128(M) x 128(N); K step 64.
// Wave (wm 0..3, wn 0..1) owns 32x64 = 2x4 C fragments; 16 wmma / K-step.
// Tiles staged with async global->LDS DMA (ASYNCcnt).
// ---------------------------------------------------------------------------
template <typename OutT>
__global__ __launch_bounds__(256)
void gemm_xwt(const __bf16* __restrict__ A, const __bf16* __restrict__ W,
              OutT* __restrict__ C, int M, int N, int K) {
    __shared__ __bf16 As[128][72];   // stride 72 bf16 = 144B (16B-aligned rows)
    __shared__ __bf16 Bs[128][72];

    const int tid  = threadIdx.x;
    const int lane = tid & 31;
    const int wave = tid >> 5;
    const int wm   = wave >> 1;   // 0..3  -> 32-row strip
    const int wn   = wave & 1;    // 0..1  -> 64-col strip
    const int m0   = blockIdx.y * 128;
    const int n0   = blockIdx.x * 128;

    const v8f vzero = {0.f,0.f,0.f,0.f,0.f,0.f,0.f,0.f};
    v8f acc[2][4];
    #pragma unroll
    for (int i = 0; i < 2; ++i)
        #pragma unroll
        for (int j = 0; j < 4; ++j) acc[i][j] = vzero;

    for (int k0 = 0; k0 < K; k0 += 64) {
        // async DMA: A and W tiles (128 x 64 bf16 each) straight into LDS
        #pragma unroll
        for (int t = tid; t < 128 * 8; t += 256) {
            const int r = t >> 3, c = (t & 7) * 8;
            async_copy_b128(&As[r][c], &A[(size_t)(m0 + r) * K + k0 + c]);
            async_copy_b128(&Bs[r][c], &W[(size_t)(n0 + r) * K + k0 + c]);
        }
        async_wait_all();
        __syncthreads();

        #pragma unroll
        for (int ks = 0; ks < 2; ++ks) {      // two 32-wide K slices
            v16bf af[2], bf[4];
            #pragma unroll
            for (int i = 0; i < 2; ++i)
                af[i] = load_frag(&As[wm * 32 + i * 16][ks * 32], 72, lane);
            #pragma unroll
            for (int j = 0; j < 4; ++j)
                bf[j] = load_frag(&Bs[wn * 64 + j * 16][ks * 32], 72, lane);
            #pragma unroll
            for (int i = 0; i < 2; ++i)
                #pragma unroll
                for (int j = 0; j < 4; ++j)
                    acc[i][j] = wmma_bf16(af[i], bf[j], acc[i][j]);
        }
        __syncthreads();
    }

    // C frag layout: VGPR r, lanes 0-15 -> (M=r, N=lane); lanes 16-31 -> (M=r+8)
    const int half = lane >> 4;
    const int col  = lane & 15;
    #pragma unroll
    for (int i = 0; i < 2; ++i) {
        #pragma unroll
        for (int j = 0; j < 4; ++j) {
            OutT* p = C + (size_t)(m0 + wm * 32 + i * 16 + 8 * half) * N
                        + (n0 + wn * 64 + j * 16 + col);
            #pragma unroll
            for (int r = 0; r < 8; ++r) {
                p[0] = (OutT)acc[i][j][r];
                p += N;
            }
        }
    }
}

// ---------------------------------------------------------------------------
// Flash attention: per block = (q-tile of 64 rows, head h, batch b).
// 128 thr = 4 waves; each wave owns 16 q rows. Online softmax.
// Q/K/V are [B*S, D] bf16, head h occupies cols [h*64, h*64+64).
// Q/K tiles staged via async global->LDS DMA; V staged transposed manually.
// ---------------------------------------------------------------------------
__global__ __launch_bounds__(128)
void attn_fused(const __bf16* __restrict__ Qg, const __bf16* __restrict__ Kg,
                const __bf16* __restrict__ Vg, __bf16* __restrict__ Og) {
    const int qt   = blockIdx.x;   // 0..31
    const int h    = blockIdx.y;   // 0..15
    const int b    = blockIdx.z;   // 0..1
    const int tid  = threadIdx.x;
    const int lane = tid & 31;
    const int wave = tid >> 5;     // 0..3
    const int half = lane >> 4;
    const int col  = lane & 15;

    __shared__ __bf16 Qs[64][72];  // stride 72 bf16 = 144B (16B aligned)
    __shared__ __bf16 Ks[64][72];
    __shared__ __bf16 Vt[64][72];  // transposed: Vt[dh][kv]
    __shared__ __bf16 Ps[64][72];  // softmax probs staging (bf16)

    const size_t rowBase = (size_t)(b * SS) * DD + (size_t)h * DH;
    const __bf16* Qb = Qg + rowBase;
    const __bf16* Kb = Kg + rowBase;
    const __bf16* Vb = Vg + rowBase;

    // ---- async-load Q tile (64 x 64) and pull this wave's A fragments ----
    for (int t = tid; t < 64 * 8; t += 128) {
        const int r = t >> 3, c = (t & 7) * 8;
        async_copy_b128(&Qs[r][c], &Qb[(size_t)(qt * 64 + r) * DD + c]);
    }
    async_wait_all();
    __syncthreads();
    v16bf qf[2];
    qf[0] = load_frag(&Qs[wave * 16][0],  72, lane);
    qf[1] = load_frag(&Qs[wave * 16][32], 72, lane);
    __syncthreads();

    const v8f vzero = {0.f,0.f,0.f,0.f,0.f,0.f,0.f,0.f};
    v8f o[4];
    o[0] = vzero; o[1] = vzero; o[2] = vzero; o[3] = vzero;
    float mx[8], sm[8];
    #pragma unroll
    for (int r = 0; r < 8; ++r) { mx[r] = -1e30f; sm[r] = 0.f; }

    for (int kt = 0; kt < SS / 64; ++kt) {
        __syncthreads();  // previous tile fully consumed
        // stage K tile via async DMA; V tile transposed through VGPRs
        for (int t = tid; t < 64 * 8; t += 128) {
            const int r = t >> 3, c = (t & 7) * 8;
            async_copy_b128(&Ks[r][c], &Kb[(size_t)(kt * 64 + r) * DD + c]);
        }
        for (int t = tid; t < 64 * 8; t += 128) {
            const int r = t >> 3, cg = t & 7;
            v8bf vv = *(const v8bf*)&Vb[(size_t)(kt * 64 + r) * DD + cg * 8];
            #pragma unroll
            for (int j = 0; j < 8; ++j) Vt[cg * 8 + j][r] = vv[j];
        }
        async_wait_all();
        __syncthreads();

        // ---- scores S = Q . K^T  (4 frags of 16x16, Dh=64 -> 2 wmma each) ----
        v8f s[4];
        #pragma unroll
        for (int j = 0; j < 4; ++j) {
            v16bf kb0 = load_frag(&Ks[j * 16][0],  72, lane);
            v16bf kb1 = load_frag(&Ks[j * 16][32], 72, lane);
            s[j] = wmma_bf16(qf[0], kb0, vzero);
            s[j] = wmma_bf16(qf[1], kb1, s[j]);
        }

        // ---- online softmax (row r lives in lanes of one 16-lane half) ----
        #pragma unroll
        for (int r = 0; r < 8; ++r) {
            float rowmax = -1e30f;
            #pragma unroll
            for (int j = 0; j < 4; ++j) rowmax = fmaxf(rowmax, s[j][r]);
            #pragma unroll
            for (int off = 8; off > 0; off >>= 1)
                rowmax = fmaxf(rowmax, __shfl_xor(rowmax, off, 32));
            const float mnew  = fmaxf(mx[r], rowmax * ATT_SCALE);
            const float alpha = __expf(mx[r] - mnew);
            float rsum = 0.f;
            #pragma unroll
            for (int j = 0; j < 4; ++j) {
                const float p = __expf(s[j][r] * ATT_SCALE - mnew);
                s[j][r] = p;
                rsum += p;
            }
            #pragma unroll
            for (int off = 8; off > 0; off >>= 1)
                rsum += __shfl_xor(rsum, off, 32);
            sm[r] = sm[r] * alpha + rsum;
            mx[r] = mnew;
            #pragma unroll
            for (int jd = 0; jd < 4; ++jd) o[jd][r] *= alpha;
            // stage P as bf16 in A-operand-consumable tile
            const int prow = wave * 16 + r + 8 * half;
            #pragma unroll
            for (int j = 0; j < 4; ++j)
                Ps[prow][j * 16 + col] = (__bf16)s[j][r];
        }
        __syncthreads();

        // ---- O += P . V  (A = P 16x64 kv, B = Vt rows = dh, kv contiguous) ----
        v16bf pA0 = load_frag(&Ps[wave * 16][0],  72, lane);
        v16bf pA1 = load_frag(&Ps[wave * 16][32], 72, lane);
        #pragma unroll
        for (int jd = 0; jd < 4; ++jd) {
            v16bf vb0 = load_frag(&Vt[jd * 16][0],  72, lane);
            v16bf vb1 = load_frag(&Vt[jd * 16][32], 72, lane);
            o[jd] = wmma_bf16(pA0, vb0, o[jd]);
            o[jd] = wmma_bf16(pA1, vb1, o[jd]);
        }
    }

    // ---- epilogue: normalize and store bf16 attn output [B*S, D] ----
    #pragma unroll
    for (int r = 0; r < 8; ++r) {
        const float inv = 1.f / sm[r];
        const int q = qt * 64 + wave * 16 + r + 8 * half;
        const size_t off = (size_t)(b * SS + q) * DD + (size_t)h * DH;
        #pragma unroll
        for (int jd = 0; jd < 4; ++jd)
            Og[off + jd * 16 + col] = (__bf16)(o[jd][r] * inv);
    }
}

// ---------------------------------------------------------------------------
// Host-side launch
// ---------------------------------------------------------------------------
extern "C" void kernel_launch(void* const* d_in, const int* in_sizes, int n_in,
                              void* d_out, int out_size, void* d_ws, size_t ws_size,
                              hipStream_t stream) {
    (void)in_sizes; (void)n_in; (void)out_size; (void)ws_size;

    const float* q_f  = (const float*)d_in[0];
    const float* Wq_f = (const float*)d_in[1];
    const float* Wk_f = (const float*)d_in[2];
    const float* Wv_f = (const float*)d_in[3];
    const float* Wo_f = (const float*)d_in[4];
    float* out = (float*)d_out;

    // workspace carve-up (bf16): X(4M) W*(4x1M) Q/K/V(3x4M) attn(4M) = 24M elems
    __bf16* Xb  = (__bf16*)d_ws;
    __bf16* Wqb = Xb  + (size_t)MROWS * DD;
    __bf16* Wkb = Wqb + (size_t)DD * DD;
    __bf16* Wvb = Wkb + (size_t)DD * DD;
    __bf16* Wob = Wvb + (size_t)DD * DD;
    __bf16* Qb  = Wob + (size_t)DD * DD;
    __bf16* Kb  = Qb  + (size_t)MROWS * DD;
    __bf16* Vb  = Kb  + (size_t)MROWS * DD;
    __bf16* Ab  = Vb  + (size_t)MROWS * DD;

    // 1) f32 -> bf16 (vectorized x4)
    cvt_f32_bf16<<<1024, 256, 0, stream>>>(q_f,  Xb,  MROWS * DD / 4);
    cvt_f32_bf16<<<256,  256, 0, stream>>>(Wq_f, Wqb, DD * DD / 4);
    cvt_f32_bf16<<<256,  256, 0, stream>>>(Wk_f, Wkb, DD * DD / 4);
    cvt_f32_bf16<<<256,  256, 0, stream>>>(Wv_f, Wvb, DD * DD / 4);
    cvt_f32_bf16<<<256,  256, 0, stream>>>(Wo_f, Wob, DD * DD / 4);

    // 2) projections Q/K/V = X . W^T (bf16 out)
    dim3 gg(DD / 128, MROWS / 128);
    gemm_xwt<__bf16><<<gg, 256, 0, stream>>>(Xb, Wqb, Qb, MROWS, DD, DD);
    gemm_xwt<__bf16><<<gg, 256, 0, stream>>>(Xb, Wkb, Kb, MROWS, DD, DD);
    gemm_xwt<__bf16><<<gg, 256, 0, stream>>>(Xb, Wvb, Vb, MROWS, DD, DD);

    // 3) fused attention per (q-tile, head, batch)
    dim3 ga(SS / 64, HH, BB);
    attn_fused<<<ga, 128, 0, stream>>>(Qb, Kb, Vb, Ab);

    // 4) output projection (f32 out to d_out)
    gemm_xwt<float><<<gg, 256, 0, stream>>>(Ab, Wob, out, MROWS, DD, DD);
}
